// LSTMModel_41300405518569
// MI455X (gfx1250) — matrix operand order repaired and checked by reading, not compiled, not run on hardware
//
#include <hip/hip_runtime.h>

typedef float v8f  __attribute__((ext_vector_type(8)));
typedef __bf16 v16bf __attribute__((ext_vector_type(16)));
typedef __bf16 bf8v  __attribute__((ext_vector_type(8)));

namespace {
enum : int {
  Bb = 16, Pp = 128, LMAXc = 128, Tt = 16384, Ff = 64, Hc = 256, Oo = 32,
  G4 = 4 * Hc,              // 1024 gate columns
  K0 = Ff + Hc,             // 320  (layer0: [x | h0])
  K1 = Hc + Hc,             // 512  (layer1: [y0 | h1])
  NC0 = K0 / 32,            // 10 bf16 K-chunks
  NC1 = K1 / 32,            // 16
  MR = 16,                  // sequences (rows) per workgroup
  P0B = K0 + 8,             // bf16 plane pitch (mult of 8 -> 16B aligned runs)
  P1B = K1 + 8,
  PG = G4 + 2,              // f32 gates pitch
  FC0 = NC0 * 2 * G4,       // 20480 B-fragments (layer 0)
  FC1 = NC1 * 2 * G4,       // 32768 B-fragments (layer 1)
};
constexpr size_t S0   = (size_t)Bb * Tt * Oo;   // out_final floats = 8,388,608
constexpr size_t HCSZ = (size_t)Bb * Pp * Hc;   // 524,288 floats per (layer, h/c)
}

// ---------------------------------------------------------------- starts scan
__global__ void k_starts(const int* __restrict__ lpp, int* __restrict__ starts) {
  int b = threadIdx.x;
  if (b < Bb) {
    int cum = 0;
    for (int p = 0; p < Pp; ++p) {
      starts[b * Pp + p] = cum;
      cum += lpp[b * Pp + p];
    }
  }
}

// ------------------------------- weight split (hi/lo bf16) + fragment packing
// B-fragment (16x16x32 bf16) per-lane order: lane half h holds K = kb+16h+e,
// fragment index = (chunk*2 + half)*1024 + n, 16 bf16 each.
__global__ void k_prep(const float* __restrict__ Wih0, const float* __restrict__ Whh0,
                       const float* __restrict__ bih0, const float* __restrict__ bhh0,
                       const float* __restrict__ Wih1, const float* __restrict__ Whh1,
                       const float* __restrict__ bih1, const float* __restrict__ bhh1,
                       __bf16* __restrict__ WB0hi, __bf16* __restrict__ WB0lo,
                       __bf16* __restrict__ WB1hi, __bf16* __restrict__ WB1lo,
                       float* __restrict__ b0, float* __restrict__ b1) {
  int idx = blockIdx.x * 256 + threadIdx.x;
  if (idx < FC0) {
    int c = idx >> 11, half = (idx >> 10) & 1, n = idx & (G4 - 1);
    int kb = c * 32 + half * 16;
    __bf16* ph = WB0hi + (size_t)idx * 16;
    __bf16* pl = WB0lo + (size_t)idx * 16;
    for (int e = 0; e < 16; ++e) {
      int k = kb + e;
      float v = (k < Ff) ? Wih0[n * Ff + k] : Whh0[n * Hc + (k - Ff)];
      __bf16 h = (__bf16)v;
      ph[e] = h;
      pl[e] = (__bf16)(v - (float)h);
    }
  } else if (idx < FC0 + FC1) {
    int i2 = idx - FC0;
    int c = i2 >> 11, half = (i2 >> 10) & 1, n = i2 & (G4 - 1);
    int kb = c * 32 + half * 16;
    __bf16* ph = WB1hi + (size_t)i2 * 16;
    __bf16* pl = WB1lo + (size_t)i2 * 16;
    for (int e = 0; e < 16; ++e) {
      int k = kb + e;
      float v = (k < Hc) ? Wih1[n * Hc + k] : Whh1[n * Hc + (k - Hc)];
      __bf16 h = (__bf16)v;
      ph[e] = h;
      pl[e] = (__bf16)(v - (float)h);
    }
  } else if (idx < FC0 + FC1 + G4) {
    int j = idx - FC0 - FC1;
    b0[j] = bih0[j] + bhh0[j];
  } else if (idx < FC0 + FC1 + 2 * G4) {
    int j = idx - FC0 - FC1 - G4;
    b1[j] = bih1[j] + bhh1[j];
  }
}

// ---------------------------------------------------------------- output zero
__global__ void k_zero(float* __restrict__ out) {
  size_t i = ((size_t)blockIdx.x * blockDim.x + threadIdx.x) * 4;
  if (i < S0) {
    float4 z = {0.f, 0.f, 0.f, 0.f};
    *(float4*)(out + i) = z;
  }
}

__device__ __forceinline__ v16bf cat8(bf8v a, bf8v b) {
  return __builtin_shufflevector(a, b, 0, 1, 2, 3, 4, 5, 6, 7,
                                 8, 9, 10, 11, 12, 13, 14, 15);
}
__device__ __forceinline__ v16bf ld16(const __bf16* p) {   // 2x b128
  return cat8(*(const bf8v*)p, *(const bf8v*)(p + 8));
}
__device__ __forceinline__ float sigmf(float x) { return 1.f / (1.f + __expf(-x)); }

// --------------------------------------- bf16x3 (f32-emulated) gate GEMM, WMMA
// gates[m][n] = sum_k A[m][k]*W^T[k][n] + bvec[n];  A ~ Ahi+Alo, W ~ Whi+Wlo.
__device__ __forceinline__ void gemm_gates(const __bf16* Ahi, const __bf16* Alo,
                                           int pitch,
                                           const __bf16* __restrict__ WBhi,
                                           const __bf16* __restrict__ WBlo,
                                           int NC, const float* __restrict__ bvec,
                                           float* gates, int colBase, int mA,
                                           int half, int nB, int mC) {
  v8f C[8];
#pragma unroll
  for (int tI = 0; tI < 8; ++tI) {
    float bb = bvec[colBase + tI * 16 + nB];
    C[tI] = (v8f){bb, bb, bb, bb, bb, bb, bb, bb};
  }
  for (int c = 0; c < NC; ++c) {
    // A-fragment: lane-half h needs K = c*32+16h+{0..15}
    const int kb = c * 32 + half * 16;
    v16bf Ah = cat8(*(const bf8v*)(Ahi + mA * pitch + kb),
                    *(const bf8v*)(Ahi + mA * pitch + kb + 8));
    v16bf Al = cat8(*(const bf8v*)(Alo + mA * pitch + kb),
                    *(const bf8v*)(Alo + mA * pitch + kb + 8));
    const size_t fb = ((size_t)(c * 2 + half) * G4 + colBase + nB) * 16;
    const __bf16* bh = WBhi + fb;
    const __bf16* bl = WBlo + fb;
#pragma unroll
    for (int tI = 0; tI < 8; ++tI) {
      v16bf Bh = ld16(bh + tI * 256);
      v16bf Bl = ld16(bl + tI * 256);
      C[tI] = __builtin_amdgcn_wmma_f32_16x16x32_bf16(
          false, Ah, false, Bh, (short)0, C[tI], false, false);
      C[tI] = __builtin_amdgcn_wmma_f32_16x16x32_bf16(
          false, Al, false, Bh, (short)0, C[tI], false, false);
      C[tI] = __builtin_amdgcn_wmma_f32_16x16x32_bf16(
          false, Ah, false, Bl, (short)0, C[tI], false, false);
    }
  }
#pragma unroll
  for (int tI = 0; tI < 8; ++tI) {
    const int n = colBase + tI * 16 + nB;
#pragma unroll
    for (int v = 0; v < 8; ++v) gates[(mC + v) * PG + n] = C[tI][v];
  }
}

// ------------------------------------------------------------- persistent LSTM
__global__ __launch_bounds__(256) void k_lstm(
    const float* __restrict__ x, const int* __restrict__ lpp,
    const __bf16* __restrict__ WB0hi, const __bf16* __restrict__ WB0lo,
    const __bf16* __restrict__ WB1hi, const __bf16* __restrict__ WB1lo,
    const float* __restrict__ b0, const float* __restrict__ b1,
    const float* __restrict__ Wfc, const float* __restrict__ bfc,
    const int* __restrict__ starts, float* __restrict__ out) {
  __shared__ __bf16 A0hi[MR * P0B], A0lo[MR * P0B];  // [x_t | h0] hi/lo planes
  __shared__ __bf16 A1hi[MR * P1B], A1lo[MR * P1B];  // [y0  | h1] hi/lo planes
  __shared__ float gates[MR * PG];   // gate exchange; y1 overlaid in cols [0,256)
  __shared__ float c0s[MR * Hc];
  __shared__ float c1s[MR * Hc];
  __shared__ int lenL[MR], startL[MR], batL[MR];

  const int tid = threadIdx.x;
  const int wave = tid >> 5, lane = tid & 31;
  const int n0 = blockIdx.x * MR;

  if (tid < MR) {
    int n = n0 + tid;
    lenL[tid]   = lpp[n];
    startL[tid] = starts[n];
    batL[tid]   = n >> 7;   // n / P
  }
  for (int i = tid; i < MR * P0B; i += 256) { A0hi[i] = (__bf16)0.f; A0lo[i] = (__bf16)0.f; }
  for (int i = tid; i < MR * P1B; i += 256) { A1hi[i] = (__bf16)0.f; A1lo[i] = (__bf16)0.f; }
  for (int i = tid; i < MR * Hc; i += 256) { c0s[i] = 0.f; c1s[i] = 0.f; }
  __syncthreads();

  const int colBase = wave * 128;
  const int mA   = lane & 15;          // A-frag row
  const int half = lane >> 4;          // A/B-frag K half
  const int nB   = lane & 15;          // B/C-frag column
  const int mC   = (lane >> 4) * 8;    // C-frag row base

  for (int t = 0; t < LMAXc; ++t) {
    {  // stage x_t tile (gathered, pos clamped like reference pos_c), split hi/lo
      int m = tid >> 4;
      int f0 = (tid & 15) * 4;
      int pos = startL[m] + t;
      if (pos > Tt - 1) pos = Tt - 1;
      const float* xr = x + ((size_t)batL[m] * Tt + pos) * Ff + f0;
      float4 v = *(const float4*)xr;
      float vv[4] = {v.x, v.y, v.z, v.w};
#pragma unroll
      for (int q = 0; q < 4; ++q) {
        __bf16 h = (__bf16)vv[q];
        A0hi[m * P0B + f0 + q] = h;
        A0lo[m * P0B + f0 + q] = (__bf16)(vv[q] - (float)h);
      }
    }
    __syncthreads();

    // ---- layer 0: gates = [x|h0] @ W0^T + b0 ----
    gemm_gates(A0hi, A0lo, P0B, WB0hi, WB0lo, NC0, b0, gates,
               colBase, mA, half, nB, mC);
    __syncthreads();

    for (int c = tid; c < MR * Hc; c += 256) {  // LSTM cell, layer 0
      int m = c >> 8, j = c & 255;
      float gi = gates[m * PG + j];
      float gf = gates[m * PG + 256 + j];
      float gg = gates[m * PG + 512 + j];
      float go = gates[m * PG + 768 + j];
      float i_ = sigmf(gi), f_ = sigmf(gf), o_ = sigmf(go);
      float g_ = tanhf(gg);
      float cn = f_ * c0s[c] + i_ * g_;
      float hn = o_ * tanhf(cn);
      bool msk = (t < lenL[m]);
      if (msk) {
        c0s[c] = cn;
        __bf16 hh = (__bf16)hn;
        A0hi[m * P0B + Ff + j] = hh;
        A0lo[m * P0B + Ff + j] = (__bf16)(hn - (float)hh);
      }
      float y0 = msk ? hn : 0.f;                 // y0 feeds layer 1
      __bf16 yh = (__bf16)y0;
      A1hi[m * P1B + j] = yh;
      A1lo[m * P1B + j] = (__bf16)(y0 - (float)yh);
    }
    __syncthreads();

    // ---- layer 1: gates = [y0|h1] @ W1^T + b1 ----
    gemm_gates(A1hi, A1lo, P1B, WB1hi, WB1lo, NC1, b1, gates,
               colBase, mA, half, nB, mC);
    __syncthreads();

    for (int c = tid; c < MR * Hc; c += 256) {  // LSTM cell, layer 1
      int m = c >> 8, j = c & 255;
      float gi = gates[m * PG + j];
      float gf = gates[m * PG + 256 + j];
      float gg = gates[m * PG + 512 + j];
      float go = gates[m * PG + 768 + j];
      float i_ = sigmf(gi), f_ = sigmf(gf), o_ = sigmf(go);
      float g_ = tanhf(gg);
      float cn = f_ * c1s[c] + i_ * g_;
      float hn = o_ * tanhf(cn);
      bool msk = (t < lenL[m]);
      if (msk) {
        c1s[c] = cn;
        __bf16 hh = (__bf16)hn;
        A1hi[m * P1B + Hc + j] = hh;
        A1lo[m * P1B + Hc + j] = (__bf16)(hn - (float)hh);
      }
      gates[m * PG + j] = msk ? hn : 0.f;        // y1 (f32) overlaid for FC
    }
    __syncthreads();

    {  // FC head + scatter (positions disjoint per packed path -> plain store)
      int m  = tid & 15;
      int oh = tid >> 4;
      if (t < lenL[m]) {
        int pos = startL[m] + t;
        size_t base = ((size_t)batL[m] * Tt + pos) * Oo;
#pragma unroll
        for (int q = 0; q < 2; ++q) {
          int o = oh + q * 16;
          float acc = bfc[o];
          const float* wr = Wfc + o * Hc;
          for (int h = 0; h < Hc; h += 2) {
            float2 y = *(const float2*)&gates[m * PG + h];
            acc += y.x * wr[h] + y.y * wr[h + 1];
          }
          out[base + o] = acc;
        }
      }
    }
    __syncthreads();
  }

  // final (h,c) states:  h_t[2,N,H] then c_t[2,N,H] after out_final
  for (int c = tid; c < MR * Hc; c += 256) {
    int m = c >> 8, j = c & 255;
    size_t n = (size_t)(n0 + m);
    float h0 = (float)A0hi[m * P0B + Ff + j] + (float)A0lo[m * P0B + Ff + j];
    float h1 = (float)A1hi[m * P1B + Hc + j] + (float)A1lo[m * P1B + Hc + j];
    out[S0 + n * Hc + j]            = h0;
    out[S0 + HCSZ + n * Hc + j]     = h1;
    out[S0 + 2 * HCSZ + n * Hc + j] = c0s[c];
    out[S0 + 3 * HCSZ + n * Hc + j] = c1s[c];
  }
}

// ----------------------------------------------------------------------------
extern "C" void kernel_launch(void* const* d_in, const int* in_sizes, int n_in,
                              void* d_out, int out_size, void* d_ws, size_t ws_size,
                              hipStream_t stream) {
  const float* x    = (const float*)d_in[0];
  const int*   lpp  = (const int*)d_in[2];
  const float* Wih0 = (const float*)d_in[4];
  const float* Whh0 = (const float*)d_in[5];
  const float* bih0 = (const float*)d_in[6];
  const float* bhh0 = (const float*)d_in[7];
  const float* Wih1 = (const float*)d_in[8];
  const float* Whh1 = (const float*)d_in[9];
  const float* bih1 = (const float*)d_in[10];
  const float* bhh1 = (const float*)d_in[11];
  const float* Wfc  = (const float*)d_in[12];
  const float* bfc  = (const float*)d_in[13];
  float* out = (float*)d_out;

  // workspace layout
  char* ws = (char*)d_ws;
  int*    starts = (int*)ws;                                    // 2048 ints
  __bf16* WB0hi = (__bf16*)(ws + 8192);                         // FC0*16 bf16
  __bf16* WB0lo = WB0hi + (size_t)FC0 * 16;
  __bf16* WB1hi = WB0lo + (size_t)FC0 * 16;                     // FC1*16 bf16
  __bf16* WB1lo = WB1hi + (size_t)FC1 * 16;
  float*  b0 = (float*)(WB1lo + (size_t)FC1 * 16);
  float*  b1 = b0 + G4;

  k_starts<<<1, 32, 0, stream>>>(lpp, starts);

  int prepTot = FC0 + FC1 + 2 * G4;
  k_prep<<<(prepTot + 255) / 256, 256, 0, stream>>>(Wih0, Whh0, bih0, bhh0,
                                                    Wih1, Whh1, bih1, bhh1,
                                                    WB0hi, WB0lo, WB1hi, WB1lo,
                                                    b0, b1);

  k_zero<<<(int)(S0 / 4 / 256), 256, 0, stream>>>(out);

  k_lstm<<<(Bb * Pp) / MR, 256, 0, stream>>>(x, lpp, WB0hi, WB0lo, WB1hi, WB1lo,
                                             b0, b1, Wfc, bfc, starts, out);
}
